// Transformer_26199300505813
// MI455X (gfx1250) — matrix operand order repaired and checked by reading, not compiled, not run on hardware
//
#include <hip/hip_runtime.h>
#include <hip/hip_bf16.h>
#include <cstdint>

// ---------------------------------------------------------------------------
// Types for WMMA fragments (wave32, v_wmma_f32_16x16x32_f16)
// ---------------------------------------------------------------------------
typedef _Float16 h2  __attribute__((ext_vector_type(2)));
typedef _Float16 h4  __attribute__((ext_vector_type(4)));
typedef _Float16 h8  __attribute__((ext_vector_type(8)));
typedef _Float16 h16 __attribute__((ext_vector_type(16)));
typedef float    f8  __attribute__((ext_vector_type(8)));

#define DIMC   1024
#define SEQ    2048
#define NROWS  4096      // 2 batches * 2048
#define NHEADS 16
#define DHEAD  64
#define MLPD   4096
#define QKVN   3072
#define NBH    32        // 2 * 16
#define ATT_SCALE 0.125f // 64^-0.5

static __device__ __forceinline__ h16 hcat(h8 lo, h8 hi) {
  return __builtin_shufflevector(lo, hi, 0,1,2,3,4,5,6,7,8,9,10,11,12,13,14,15);
}
// A fragment (16x32 f16): lane L<16 holds row L, K {0..7,16..23}; L>=16: K {8..15,24..31}
static __device__ __forceinline__ h16 fragA(const _Float16* base, int stride, int koff, int lane) {
  const _Float16* p = base + (size_t)(lane & 15) * stride + koff + ((lane >> 4) << 3);
  return hcat(*(const h8*)p, *(const h8*)(p + 16));
}
// B fragment (32x16 f16) read from Bt[N][K] rows: lane L<16 holds col L, K {0..7,16..23}; L>=16: K {8..15,24..31}
static __device__ __forceinline__ h16 fragB(const _Float16* base, int stride, int koff, int lane) {
  const _Float16* p = base + (size_t)(lane & 15) * stride + koff + ((lane >> 4) << 3);
  return hcat(*(const h8*)p, *(const h8*)(p + 16));
}
static __device__ __forceinline__ f8 wmma32(h16 a, h16 b, f8 c) {
  return __builtin_amdgcn_wmma_f32_16x16x32_f16(false, a, false, b, (short)0, c, false, false);
}

// ---------------------------------------------------------------------------
// CDNA5 async global->LDS DMA (ASYNCcnt-tracked), 16B per lane per issue.
// LDS operand is the wave-relative byte offset = low 32 bits of the flat addr
// (ISA: flat LDS aperture maps addr[31:0] -> LDS offset).
// ---------------------------------------------------------------------------
static __device__ __forceinline__ void async_load16(void* lds, const void* gptr) {
  uint32_t off = (uint32_t)(uintptr_t)lds;
  asm volatile("global_load_async_to_lds_b128 %0, %1, off"
               :: "v"(off), "v"(gptr)
               : "memory");
}
static __device__ __forceinline__ void wait_async0() {
  asm volatile("s_wait_asynccnt 0x0" ::: "memory");
}

// ---------------------------------------------------------------------------
// Weight convert: in f32 [K][N] -> out f16 [N][K]  (transposed, per grid.z slab)
// ---------------------------------------------------------------------------
__global__ __launch_bounds__(256) void k_transpose_cvt(const float* __restrict__ in,
                                                       _Float16* __restrict__ out,
                                                       int K, int N) {
  const size_t mat = (size_t)K * N;
  in  += (size_t)blockIdx.z * mat;
  out += (size_t)blockIdx.z * mat;
  __shared__ float t[32][33];
  const int n0 = blockIdx.x * 32, k0 = blockIdx.y * 32;
  const int tx = threadIdx.x, ty = threadIdx.y;
#pragma unroll
  for (int i = 0; i < 4; ++i) {
    int k = ty + i * 8;
    t[k][tx] = in[(size_t)(k0 + k) * N + n0 + tx];
  }
  __syncthreads();
#pragma unroll
  for (int i = 0; i < 4; ++i) {
    int n = ty + i * 8;
    out[(size_t)(n0 + n) * K + k0 + tx] = (_Float16)t[tx][n];
  }
}

// ---------------------------------------------------------------------------
// LayerNorm row (1024) -> f16 out. One block per row, 256 threads x float4.
// ---------------------------------------------------------------------------
__global__ __launch_bounds__(256) void k_layernorm(const float* __restrict__ x,
                                                   const float* __restrict__ g,
                                                   const float* __restrict__ b,
                                                   _Float16* __restrict__ out) {
  const int row = blockIdx.x, tid = threadIdx.x;
  const float4 v = ((const float4*)(x + (size_t)row * DIMC))[tid];
  float s  = v.x + v.y + v.z + v.w;
  float s2 = v.x * v.x + v.y * v.y + v.z * v.z + v.w * v.w;
#pragma unroll
  for (int m = 1; m < 32; m <<= 1) {
    s  += __shfl_xor(s,  m, 32);
    s2 += __shfl_xor(s2, m, 32);
  }
  __shared__ float rs[8], rs2[8];
  if ((tid & 31) == 0) { rs[tid >> 5] = s; rs2[tid >> 5] = s2; }
  __syncthreads();
  float S = 0.f, S2 = 0.f;
#pragma unroll
  for (int i = 0; i < 8; ++i) { S += rs[i]; S2 += rs2[i]; }
  const float mean = S * (1.f / DIMC);
  const float var  = S2 * (1.f / DIMC) - mean * mean;
  const float rstd = rsqrtf(var + 1e-5f);
  const float4 gv = ((const float4*)g)[tid];
  const float4 bv = ((const float4*)b)[tid];
  h4 o;
  o[0] = (_Float16)((v.x - mean) * rstd * gv.x + bv.x);
  o[1] = (_Float16)((v.y - mean) * rstd * gv.y + bv.y);
  o[2] = (_Float16)((v.z - mean) * rstd * gv.z + bv.z);
  o[3] = (_Float16)((v.w - mean) * rstd * gv.w + bv.w);
  ((h4*)(out + (size_t)row * DIMC))[tid] = o;
}

// ---------------------------------------------------------------------------
// WMMA GEMM: C[M,N] = A[M,K](f16,row) @ Bt[N,K](f16,row) (+bias)(+res)(gelu?)
// Block tile 128x256, 8 waves each 64x64 (16 wmma accumulators).
// Double-buffered LDS fed by global_load_async_to_lds_b128.
// ---------------------------------------------------------------------------
template <bool GELU, bool OUTF16>
__global__ __launch_bounds__(256) void k_gemm(const _Float16* __restrict__ A,
                                              const _Float16* __restrict__ Bt,
                                              const float* __restrict__ bias,
                                              const float* res,
                                              float* Cf32, _Float16* Cf16,
                                              int M, int N, int K) {
  const int tid = threadIdx.x, lane = tid & 31, w = tid >> 5;
  const int wm = w >> 2, wn = w & 3;
  const int m0 = blockIdx.y * 128, n0 = blockIdx.x * 256;
  __shared__ _Float16 As[2][128][40];
  __shared__ _Float16 Bs[2][256][40];

  f8 acc[4][4];
#pragma unroll
  for (int mi = 0; mi < 4; ++mi)
#pragma unroll
    for (int ni = 0; ni < 4; ++ni) acc[mi][ni] = (f8)(0.0f);

  auto issue_stage = [&](int k0, int buf) {
#pragma unroll
    for (int i = tid; i < 512; i += 256) {         // A tile: 128 rows x 4 h8-chunks
      int r = i >> 2, c = i & 3;
      async_load16(&As[buf][r][c * 8], A + (size_t)(m0 + r) * K + k0 + c * 8);
    }
#pragma unroll
    for (int i = tid; i < 1024; i += 256) {        // Bt tile: 256 rows x 4 h8-chunks
      int r = i >> 2, c = i & 3;
      async_load16(&Bs[buf][r][c * 8], Bt + (size_t)(n0 + r) * K + k0 + c * 8);
    }
  };

  const int nk = K >> 5;
  issue_stage(0, 0);
  for (int ks = 0; ks < nk; ++ks) {
    wait_async0();          // stage ks resident (issued last iteration)
    __syncthreads();
    const int cur = ks & 1;
    if (ks + 1 < nk) issue_stage((ks + 1) << 5, cur ^ 1);  // overlap with compute

    h16 af[4], bf[4];
#pragma unroll
    for (int mi = 0; mi < 4; ++mi) af[mi] = fragA(&As[cur][wm * 64 + mi * 16][0], 40, 0, lane);
#pragma unroll
    for (int ni = 0; ni < 4; ++ni) bf[ni] = fragB(&Bs[cur][wn * 64 + ni * 16][0], 40, 0, lane);
#pragma unroll
    for (int mi = 0; mi < 4; ++mi)
#pragma unroll
      for (int ni = 0; ni < 4; ++ni) acc[mi][ni] = wmma32(af[mi], bf[ni], acc[mi][ni]);
  }

  // Epilogue. C layout: VGPR r holds rows {r, r+8}; cols = lane&15 per 16-lane half.
#pragma unroll
  for (int ni = 0; ni < 4; ++ni) {
    const int col = n0 + wn * 64 + ni * 16 + (lane & 15);
    const float bv = bias ? bias[col] : 0.f;
#pragma unroll
    for (int mi = 0; mi < 4; ++mi) {
      const int rb = m0 + wm * 64 + mi * 16 + ((lane >> 4) << 3);
#pragma unroll
      for (int r = 0; r < 8; ++r) {
        const int row = rb + r;
        float v = acc[mi][ni][r] + bv;
        if (res) v += res[(size_t)row * N + col];
        if (GELU) v = 0.5f * v * (1.f + erff(v * 0.70710678118654752f));
        if (OUTF16) Cf16[(size_t)row * N + col] = (_Float16)v;
        else        Cf32[(size_t)row * N + col] = v;
      }
    }
  }
}

// ---------------------------------------------------------------------------
// RoPE + pack q,k: qkv f32 [4096][3072] -> q_h,k_h f16 [bh][2048][64]
// ---------------------------------------------------------------------------
__global__ __launch_bounds__(256) void k_rope_pack(const float* __restrict__ qkv,
                                                   const float* __restrict__ inv_freq,
                                                   _Float16* __restrict__ qh,
                                                   _Float16* __restrict__ kh) {
  const int idx = blockIdx.x * 256 + threadIdx.x;   // 4096*16*32 = 2^21 threads
  const int p = idx & 31;
  const int h = (idx >> 5) & 15;
  const int row = idx >> 9;
  const int n = row & (SEQ - 1);
  const int b = row >> 11;
  const int bh = b * NHEADS + h;
  const float* src = qkv + (size_t)row * QKVN + h * DHEAD + 2 * p;
  float q1 = src[0], q2 = src[1];
  float k1 = src[DIMC], k2 = src[DIMC + 1];
  if (p < 16) {
    float fr = (float)n * inv_freq[p];
    float sn, cs;
    __sincosf(fr, &sn, &cs);
    float a = q1 * cs - q2 * sn, c = q2 * cs + q1 * sn;
    q1 = a; q2 = c;
    a = k1 * cs - k2 * sn; c = k2 * cs + k1 * sn;
    k1 = a; k2 = c;
  }
  h2 qo; qo[0] = (_Float16)q1; qo[1] = (_Float16)q2;
  h2 ko; ko[0] = (_Float16)k1; ko[1] = (_Float16)k2;
  *(h2*)(qh + ((size_t)bh * SEQ + n) * DHEAD + 2 * p) = qo;
  *(h2*)(kh + ((size_t)bh * SEQ + n) * DHEAD + 2 * p) = ko;
}

// ---------------------------------------------------------------------------
// V pack transposed: qkv f32 -> vt f16 [bh][64][2048]
// ---------------------------------------------------------------------------
__global__ __launch_bounds__(256) void k_vpack_t(const float* __restrict__ qkv,
                                                 _Float16* __restrict__ vt) {
  const int bh = blockIdx.z, b = bh >> 4, h = bh & 15;
  const int n0 = blockIdx.x * 32, d0 = blockIdx.y * 32;
  const int tx = threadIdx.x, ty = threadIdx.y;
  __shared__ float t[32][33];
#pragma unroll
  for (int i = 0; i < 4; ++i) {
    int ni = ty + i * 8;
    t[ni][tx] = qkv[(size_t)(b * SEQ + n0 + ni) * QKVN + 2 * DIMC + h * DHEAD + d0 + tx];
  }
  __syncthreads();
#pragma unroll
  for (int i = 0; i < 4; ++i) {
    int di = ty + i * 8;
    vt[((size_t)bh * DHEAD + d0 + di) * SEQ + n0 + tx] = (_Float16)t[tx][di];
  }
}

// ---------------------------------------------------------------------------
// Flash attention: per (bh, 64-query block). 4 waves x 16 rows each.
// 64-key chunks of K and Vt streamed via async DMA, double-buffered in LDS.
// ---------------------------------------------------------------------------
__global__ __launch_bounds__(128) void k_flash_attn(const _Float16* __restrict__ qh,
                                                    const _Float16* __restrict__ kh,
                                                    const _Float16* __restrict__ vt,
                                                    _Float16* __restrict__ attn_out) {
  const int bh = blockIdx.y;
  const int q0 = blockIdx.x * 64;
  const int tid = threadIdx.x;
  const int w = tid >> 5, lane = tid & 31;

  __shared__ _Float16 Ks[2][64][72];   // keys x d
  __shared__ _Float16 Vs[2][64][72];   // d x keys (Vt chunk)
  __shared__ _Float16 Ps[4][16][72];   // per-wave P scratch (D-layout -> A-layout)

  auto issue_chunk = [&](int kc, int buf) {
#pragma unroll
    for (int i = tid; i < 512; i += 128) {     // K chunk: 64 rows x 8 h8-chunks
      int r = i >> 3, c = i & 7;
      async_load16(&Ks[buf][r][c * 8], kh + ((size_t)bh * SEQ + kc + r) * DHEAD + c * 8);
    }
#pragma unroll
    for (int i = tid; i < 512; i += 128) {     // Vt chunk: 64 d-rows x 8 h8-chunks
      int r = i >> 3, c = i & 7;
      async_load16(&Vs[buf][r][c * 8], vt + ((size_t)bh * DHEAD + r) * SEQ + kc + c * 8);
    }
  };

  // Q fragments for this wave's 16 rows (d = 64 -> 2 k-steps)
  const _Float16* qbase = qh + ((size_t)bh * SEQ + q0 + w * 16) * DHEAD;
  h16 qf[2];
#pragma unroll
  for (int s = 0; s < 2; ++s) qf[s] = fragA(qbase, DHEAD, s * 32, lane);

  f8 o[4];
#pragma unroll
  for (int ni = 0; ni < 4; ++ni) o[ni] = (f8)(0.0f);
  float m[8], l[8], alpha[8];
#pragma unroll
  for (int r = 0; r < 8; ++r) { m[r] = -3.0e38f; l[r] = 0.f; }

  issue_chunk(0, 0);
  for (int kc = 0; kc < SEQ; kc += 64) {
    wait_async0();
    __syncthreads();
    const int cur = (kc >> 6) & 1;
    if (kc + 64 < SEQ) issue_chunk(kc + 64, cur ^ 1);

    // S = Q @ K^T over the 64-key chunk
    f8 s[4];
#pragma unroll
    for (int ni = 0; ni < 4; ++ni) s[ni] = (f8)(0.0f);
#pragma unroll
    for (int st = 0; st < 2; ++st)
#pragma unroll
      for (int ni = 0; ni < 4; ++ni) {
        h16 bf = fragB(&Ks[cur][ni * 16][0], 72, st * 32, lane);
        s[ni] = wmma32(qf[st], bf, s[ni]);
      }

    // online softmax (rows live in 16-lane halves matching C-fragment layout)
#pragma unroll
    for (int r = 0; r < 8; ++r) {
      float t = fmaxf(fmaxf(s[0][r], s[1][r]), fmaxf(s[2][r], s[3][r])) * ATT_SCALE;
#pragma unroll
      for (int msk = 1; msk < 16; msk <<= 1) t = fmaxf(t, __shfl_xor(t, msk, 32));
      float mn = fmaxf(m[r], t);
      alpha[r] = __expf(m[r] - mn);
      m[r] = mn;
    }
#pragma unroll
    for (int ni = 0; ni < 4; ++ni)
#pragma unroll
      for (int r = 0; r < 8; ++r) s[ni][r] = __expf(s[ni][r] * ATT_SCALE - m[r]);
#pragma unroll
    for (int r = 0; r < 8; ++r) {
      float t = s[0][r] + s[1][r] + s[2][r] + s[3][r];
#pragma unroll
      for (int msk = 1; msk < 16; msk <<= 1) t += __shfl_xor(t, msk, 32);
      l[r] = l[r] * alpha[r] + t;
    }
#pragma unroll
    for (int ni = 0; ni < 4; ++ni)
#pragma unroll
      for (int r = 0; r < 8; ++r) o[ni][r] *= alpha[r];

    // stash P to LDS (D-layout scatter), reload as A fragments
    const int rr = (lane >> 4) << 3;
#pragma unroll
    for (int ni = 0; ni < 4; ++ni)
#pragma unroll
      for (int r = 0; r < 8; ++r)
        Ps[w][r + rr][ni * 16 + (lane & 15)] = (_Float16)s[ni][r];
    __syncthreads();

    h16 pf[2];
#pragma unroll
    for (int st = 0; st < 2; ++st) pf[st] = fragA(&Ps[w][0][0], 72, st * 32, lane);
#pragma unroll
    for (int st = 0; st < 2; ++st)
#pragma unroll
      for (int ni = 0; ni < 4; ++ni) {
        h16 vf = fragB(&Vs[cur][ni * 16][0], 72, st * 32, lane);
        o[ni] = wmma32(pf[st], vf, o[ni]);
      }
  }

  // write out: attn_out[token][h*64 + d] f16
  float invl[8];
#pragma unroll
  for (int r = 0; r < 8; ++r) invl[r] = 1.f / l[r];
  const int b = bh >> 4, h = bh & 15;
#pragma unroll
  for (int ni = 0; ni < 4; ++ni) {
    const int col = h * DHEAD + ni * 16 + (lane & 15);
#pragma unroll
    for (int r = 0; r < 8; ++r) {
      const int token = b * SEQ + q0 + w * 16 + ((lane >> 4) << 3) + r;
      attn_out[(size_t)token * DIMC + col] = (_Float16)(o[ni][r] * invl[r]);
    }
  }
}

// ---------------------------------------------------------------------------
// Orchestration
// ---------------------------------------------------------------------------
extern "C" void kernel_launch(void* const* d_in, const int* in_sizes, int n_in,
                              void* d_out, int out_size, void* d_ws, size_t ws_size,
                              hipStream_t stream) {
  (void)in_sizes; (void)n_in; (void)out_size; (void)ws_size;
  const float* x        = (const float*)d_in[0];
  const float* wqkv     = (const float*)d_in[1];   // [1024][3072]
  const float* wo       = (const float*)d_in[2];   // [1024][1024]
  const float* bo       = (const float*)d_in[3];
  const float* inv_freq = (const float*)d_in[4];
  const float* ln1_g    = (const float*)d_in[5];
  const float* ln1_b    = (const float*)d_in[6];
  const float* ln2_g    = (const float*)d_in[7];
  const float* ln2_b    = (const float*)d_in[8];
  const float* w1       = (const float*)d_in[9];   // [6][1024][4096]
  const float* b1       = (const float*)d_in[10];  // [6][4096]
  const float* w2       = (const float*)d_in[11];  // [6][4096][1024]
  const float* b2       = (const float*)d_in[12];  // [6][1024]
  float* xres = (float*)d_out;                     // residual stream, f32 [4096][1024]

  char* p = (char*)d_ws;
  auto carve = [&](size_t bytes) -> void* {
    void* r = (void*)p;
    p += (bytes + 255) & ~(size_t)255;
    return r;
  };
  _Float16* wqkv_t = (_Float16*)carve((size_t)QKVN * DIMC * 2);
  _Float16* wo_t   = (_Float16*)carve((size_t)DIMC * DIMC * 2);
  _Float16* w1_t   = (_Float16*)carve((size_t)6 * MLPD * DIMC * 2);
  _Float16* w2_t   = (_Float16*)carve((size_t)6 * DIMC * MLPD * 2);
  _Float16* xn_h   = (_Float16*)carve((size_t)NROWS * DIMC * 2);
  float*    qkv_f  = (float*)   carve((size_t)NROWS * QKVN * 4);
  _Float16* q_h    = (_Float16*)carve((size_t)NBH * SEQ * DHEAD * 2);
  _Float16* k_h    = (_Float16*)carve((size_t)NBH * SEQ * DHEAD * 2);
  _Float16* v_t    = (_Float16*)carve((size_t)NBH * DHEAD * SEQ * 2);
  _Float16* attn_h = (_Float16*)carve((size_t)NROWS * DIMC * 2);
  _Float16* g_h    = (_Float16*)qkv_f;  // disjoint lifetime: reuse qkv f32 region

  // ---- per-call weight down-convert + transpose (f32 -> f16 W^T)
  k_transpose_cvt<<<dim3(QKVN / 32, DIMC / 32, 1), dim3(32, 8), 0, stream>>>(wqkv, wqkv_t, DIMC, QKVN);
  k_transpose_cvt<<<dim3(DIMC / 32, DIMC / 32, 1), dim3(32, 8), 0, stream>>>(wo, wo_t, DIMC, DIMC);
  k_transpose_cvt<<<dim3(MLPD / 32, DIMC / 32, 6), dim3(32, 8), 0, stream>>>(w1, w1_t, DIMC, MLPD);
  k_transpose_cvt<<<dim3(DIMC / 32, MLPD / 32, 6), dim3(32, 8), 0, stream>>>(w2, w2_t, MLPD, DIMC);

  // residual stream <- x
  hipMemcpyAsync(xres, x, (size_t)NROWS * DIMC * 4, hipMemcpyDeviceToDevice, stream);

  for (int lyr = 0; lyr < 6; ++lyr) {
    // --- attention block ---
    k_layernorm<<<NROWS, 256, 0, stream>>>(xres, ln1_g + lyr * DIMC, ln1_b + lyr * DIMC, xn_h);
    k_gemm<false, false><<<dim3(QKVN / 256, NROWS / 128), 256, 0, stream>>>(
        xn_h, wqkv_t, nullptr, nullptr, qkv_f, nullptr, NROWS, QKVN, DIMC);
    k_rope_pack<<<(NROWS * NHEADS * 32) / 256, 256, 0, stream>>>(qkv_f, inv_freq, q_h, k_h);
    k_vpack_t<<<dim3(SEQ / 32, DHEAD / 32, NBH), dim3(32, 8), 0, stream>>>(qkv_f, v_t);
    k_flash_attn<<<dim3(SEQ / 64, NBH), 128, 0, stream>>>(q_h, k_h, v_t, attn_h);
    k_gemm<false, false><<<dim3(DIMC / 256, NROWS / 128), 256, 0, stream>>>(
        attn_h, wo_t, bo, xres, xres, nullptr, NROWS, DIMC, DIMC);

    // --- MLP block ---
    k_layernorm<<<NROWS, 256, 0, stream>>>(xres, ln2_g + lyr * DIMC, ln2_b + lyr * DIMC, xn_h);
    k_gemm<true, true><<<dim3(MLPD / 256, NROWS / 128), 256, 0, stream>>>(
        xn_h, w1_t + (size_t)lyr * MLPD * DIMC, b1 + lyr * MLPD, nullptr,
        nullptr, g_h, NROWS, MLPD, DIMC);
    k_gemm<false, false><<<dim3(DIMC / 256, NROWS / 128), 256, 0, stream>>>(
        g_h, w2_t + (size_t)lyr * DIMC * MLPD, b2 + lyr * DIMC, xres,
        xres, nullptr, NROWS, DIMC, MLPD);
  }
}
